// NWC_lattice_69518340653328
// MI455X (gfx1250) — matrix operand order repaired and checked by reading, not compiled, not run on hardware
//
#include <hip/hip_runtime.h>

// ---------------------------------------------------------------------------
// Fused NWC lattice autoencoder forward for MI455X (gfx1250, wave32, WMMA).
//   x -> (x-shift)/scale -> encoder_a -> E8P quantize -> encoder_s -> *scale+shift
// One workgroup processes a 128-row tile end-to-end; all activations stay in
// LDS; all GEMMs use v_wmma_f32_16x16x32_bf16 with f32 accumulation.
// Each wave computes 4 output-column tiles per K-pass so one A-fragment load
// feeds 4 back-to-back WMMAs (XDL pipelining + 4x less LDS traffic).
// ---------------------------------------------------------------------------

#define DEV __device__ __forceinline__

typedef __bf16 bf16;
typedef __attribute__((ext_vector_type(16))) __bf16 v16bf;
typedef __attribute__((ext_vector_type(8)))  __bf16 bf16x8;
typedef __attribute__((ext_vector_type(8)))  float  v8f;
typedef __attribute__((ext_vector_type(4)))  float  f32x4;

static constexpr int N_ROWS = 262144;
static constexpr int D      = 128;
static constexpr int H      = 512;
static constexpr int M      = 256;
static constexpr int R      = 2;
static constexpr int TILE   = 128;             // rows per workgroup
static constexpr int BLOCK  = 256;             // 8 wave32
static constexpr int RSTRIDE = 1040;           // bytes per LDS activation row (padded)
static constexpr int SMEM_BYTES = 2 * TILE * RSTRIDE;   // 266,240 B <= 320 KB
static constexpr float LN_EPS = 1e-5f;

// ---------------------------------------------------------------------------
// f32 -> bf16 weight conversion (into workspace, re-done every launch; cheap
// and deterministic so it is graph-capture safe).
// ---------------------------------------------------------------------------
__global__ void cvt_f32_bf16(const float* __restrict__ s, bf16* __restrict__ d, int n) {
    int i = blockIdx.x * blockDim.x + threadIdx.x;
    if (i < n) d[i] = (bf16)s[i];
}

// ---------------------------------------------------------------------------
// Wave-level GEMM: each wave computes its 16 rows x outCols, grouped as
// 4 x (16x16) output tiles per pass, K-chunked by 32.
//   A (16x32 bf16) from LDS activation buffer (contiguous k order, k = 16*half + s),
//   loaded ONCE per K-chunk and reused by 4 WMMAs.
//   B (32x16 bf16) = W^T chunk; W is [outCols x K] row-major bf16 so each
//   lane's 16 K-values are 32 contiguous bytes (L2-resident weight stream).
// MODE 0: out = bf16 -> LDS (acc+bias)
// MODE 1: out = f32  -> LDS (acc+bias)            (pre-quantization y)
// MODE 2: out = f32  -> global, scale*val+shift   (final x_hat, width D)
// ---------------------------------------------------------------------------
template<int MODE>
DEV void wave_gemm(const char* inBuf, int K, char* outBuf, int outCols,
                   const bf16* __restrict__ W, const float* __restrict__ bias,
                   const float* __restrict__ scale, const float* __restrict__ shift,
                   float* __restrict__ gout, int growBase, int tid)
{
    const int lane    = tid & 31;
    const int wave    = tid >> 5;
    const int rowbase = wave * 16;
    const int half    = lane >> 4;     // 0: K-lo half, 1: K-hi half
    const int mrow    = lane & 15;     // A row within tile / B column within tile
    const char* arow  = inBuf + (size_t)(rowbase + mrow) * RSTRIDE;

    for (int n0 = 0; n0 < outCols; n0 += 64) {
        // 4 column tiles share one A fragment per K-chunk.
        const bf16* wrow = W + (size_t)(n0 + mrow) * K + half * 16;
        v8f acc[4];
#pragma unroll
        for (int t = 0; t < 4; ++t) acc[t] = (v8f){};

        for (int k0 = 0; k0 < K; k0 += 32) {
            // A fragment: lanes 0-15 hold K=[k0..k0+15] of row mrow,
            //             lanes 16-31 hold K=[k0+16..k0+31].
            bf16x8 alo = *(const bf16x8*)(arow + (size_t)(k0 + half * 16) * 2);
            bf16x8 ahi = *(const bf16x8*)(arow + (size_t)(k0 + half * 16 + 8) * 2);
            v16bf a;
#pragma unroll
            for (int j = 0; j < 8; ++j) { a[j] = alo[j]; a[j + 8] = ahi[j]; }
            __builtin_prefetch(wrow + k0 + 128, 0, 1);   // global_prefetch weight stream
#pragma unroll
            for (int t = 0; t < 4; ++t) {
                // B fragment: lane (<16) covers K=k0..k0+15 of column n0+t*16+mrow.
                v16bf b = *(const v16bf*)(wrow + (size_t)t * 16 * K + k0);
                acc[t] = __builtin_amdgcn_wmma_f32_16x16x32_bf16(
                    false, a, false, b, (short)0, acc[t], false, false);
            }
        }

#pragma unroll
        for (int t = 0; t < 4; ++t) {
            const int col = n0 + t * 16 + mrow;
            const float bv = bias[col];
            // C/D layout: VGPR i -> row (rowbase + half*8 + i), column col.
            if constexpr (MODE == 2) {
                const float sc = scale[col], sh = shift[col];
#pragma unroll
                for (int i = 0; i < 8; ++i) {
                    const int r = rowbase + half * 8 + i;
                    gout[(size_t)(growBase + r) * D + col] = sc * (acc[t][i] + bv) + sh;
                }
            } else if constexpr (MODE == 1) {
#pragma unroll
                for (int i = 0; i < 8; ++i) {
                    const int r = rowbase + half * 8 + i;
                    ((float*)(outBuf + (size_t)r * RSTRIDE))[col] = acc[t][i] + bv;
                }
            } else {
#pragma unroll
                for (int i = 0; i < 8; ++i) {
                    const int r = rowbase + half * 8 + i;
                    ((bf16*)(outBuf + (size_t)r * RSTRIDE))[col] = (bf16)(acc[t][i] + bv);
                }
            }
        }
    }
}

// ---------------------------------------------------------------------------
// h += relu(layernorm(r) * g + b), wave-local: 2 lanes per row, pair-reduce
// moments with __shfl_xor (wave32), vectorized bf16x8 LDS traffic.
// ---------------------------------------------------------------------------
DEV void ln_relu_add(char* hBuf, const char* rBuf,
                     const float* __restrict__ g, const float* __restrict__ b, int tid)
{
    const int lane    = tid & 31;
    const int wave    = tid >> 5;
    const int row     = wave * 16 + (lane >> 1);
    const int colbase = (lane & 1) * 256;
    const char* rrow  = rBuf + (size_t)row * RSTRIDE;
    char*       hrow  = hBuf + (size_t)row * RSTRIDE;

    float sum = 0.f, ss = 0.f;
    for (int c = 0; c < 256; c += 8) {
        bf16x8 rv = *(const bf16x8*)(rrow + (size_t)(colbase + c) * 2);
#pragma unroll
        for (int j = 0; j < 8; ++j) { const float f = (float)rv[j]; sum += f; ss += f * f; }
    }
    sum += __shfl_xor(sum, 1, 32);
    ss  += __shfl_xor(ss, 1, 32);
    const float mu  = sum * (1.f / (float)H);
    const float var = ss  * (1.f / (float)H) - mu * mu;
    const float inv = rsqrtf(var + LN_EPS);

    for (int c = 0; c < 256; c += 8) {
        bf16x8 rv = *(const bf16x8*)(rrow + (size_t)(colbase + c) * 2);
        bf16x8 hv = *(const bf16x8*)(hrow + (size_t)(colbase + c) * 2);
#pragma unroll
        for (int j = 0; j < 8; ++j) {
            const int col = colbase + c + j;
            float v = ((float)rv[j] - mu) * inv * g[col] + b[col];
            v = v > 0.f ? v : 0.f;
            hv[j] = (bf16)((float)hv[j] + v);
        }
        *(bf16x8*)(hrow + (size_t)(colbase + c) * 2) = hv;
    }
}

// ---------------------------------------------------------------------------
// E8P lattice quantizer (nearest point of E8 + 1/4).
// ---------------------------------------------------------------------------
DEV void round_d8(const float z[8], float f[8])
{
    float d[8];
#pragma unroll
    for (int j = 0; j < 8; ++j) { f[j] = rintf(z[j]); d[j] = z[j] - f[j]; }
    int idx = 0; float best = fabsf(d[0]);
#pragma unroll
    for (int j = 1; j < 8; ++j) { const float a = fabsf(d[j]); if (a > best) { best = a; idx = j; } }
    float s = 0.f;
#pragma unroll
    for (int j = 0; j < 8; ++j) s += f[j];
    const bool odd = fabsf(fmodf(s, 2.0f)) > 0.5f;
    if (odd) f[idx] += (d[idx] >= 0.f) ? 1.f : -1.f;
}

DEV void e8p_quant(const float y[8], float q[8])
{
    float z[8], z2[8], fa[8], fb[8];
#pragma unroll
    for (int j = 0; j < 8; ++j) { z[j] = y[j] - 0.25f; z2[j] = z[j] - 0.5f; }
    round_d8(z,  fa);
    round_d8(z2, fb);
#pragma unroll
    for (int j = 0; j < 8; ++j) fb[j] += 0.5f;
    float da = 0.f, db = 0.f;
#pragma unroll
    for (int j = 0; j < 8; ++j) {
        const float ea = z[j] - fa[j], eb = z[j] - fb[j];
        da += ea * ea; db += eb * eb;
    }
#pragma unroll
    for (int j = 0; j < 8; ++j) q[j] = ((da <= db) ? fa[j] : fb[j]) + 0.25f;
}

// ---------------------------------------------------------------------------
// Fully fused pipeline kernel: one 128-row tile per workgroup.
// ---------------------------------------------------------------------------
__global__ __launch_bounds__(BLOCK)
void fused_nwc_kernel(
    const float* __restrict__ x, const float* __restrict__ scale, const float* __restrict__ shift,
    const bf16* __restrict__ ga_w_in,  const float* __restrict__ ga_b_in,
    const bf16* __restrict__ ga_res_w, const float* __restrict__ ga_res_b,
    const float* __restrict__ ga_ln_g, const float* __restrict__ ga_ln_b,
    const bf16* __restrict__ ga_w_out, const float* __restrict__ ga_b_out,
    const bf16* __restrict__ gs_w_in,  const float* __restrict__ gs_b_in,
    const bf16* __restrict__ gs_res_w, const float* __restrict__ gs_res_b,
    const float* __restrict__ gs_ln_g, const float* __restrict__ gs_ln_b,
    const bf16* __restrict__ gs_w_out, const float* __restrict__ gs_b_out,
    float* __restrict__ out)
{
    extern __shared__ char smem[];
    char* bufA = smem;                    // TILE x RSTRIDE
    char* bufB = smem + TILE * RSTRIDE;   // TILE x RSTRIDE

    const int tid      = threadIdx.x;
    const int growBase = blockIdx.x * TILE;

    // --- stage 0: load x tile (float4), (x - shift)/scale, bf16 into bufA
    for (int idx = tid; idx < TILE * (D / 4); idx += BLOCK) {
        const int r  = idx >> 5;           // D/4 == 32 quads per row
        const int c4 = (idx & 31) * 4;
        f32x4 v = *(const f32x4*)(x + (size_t)(growBase + r) * D + c4);
        bf16* orow = (bf16*)(bufA + (size_t)r * RSTRIDE) + c4;
#pragma unroll
        for (int j = 0; j < 4; ++j) {
            orow[j] = (bf16)((v[j] - shift[c4 + j]) / scale[c4 + j]);
        }
    }
    __syncthreads();

    // --- encoder a ---------------------------------------------------------
    wave_gemm<0>(bufA, D, bufB, H, ga_w_in, ga_b_in, nullptr, nullptr, nullptr, 0, tid);
    __syncthreads();
#pragma unroll
    for (int i = 0; i < R; ++i) {
        wave_gemm<0>(bufB, H, bufA, H, ga_res_w + (size_t)i * H * H, ga_res_b + i * H,
                     nullptr, nullptr, nullptr, 0, tid);
        __syncthreads();
        ln_relu_add(bufB, bufA, ga_ln_g + i * H, ga_ln_b + i * H, tid);
        __syncthreads();
    }
    // y (f32) into bufA
    wave_gemm<1>(bufB, H, bufA, M, ga_w_out, ga_b_out, nullptr, nullptr, nullptr, 0, tid);
    __syncthreads();

    // --- E8P quantize: bufA f32 [TILE][M] -> bufB bf16 [TILE][M] -----------
    for (int gi = tid; gi < TILE * (M / 8); gi += BLOCK) {
        const int r = gi >> 5;             // M/8 == 32 groups per row
        const int q = gi & 31;
        const float* yrow = (const float*)(bufA + (size_t)r * RSTRIDE) + q * 8;
        float y8[8], qv[8];
#pragma unroll
        for (int j = 0; j < 8; ++j) y8[j] = yrow[j];
        e8p_quant(y8, qv);
        bf16* orow = (bf16*)(bufB + (size_t)r * RSTRIDE) + q * 8;
#pragma unroll
        for (int j = 0; j < 8; ++j) orow[j] = (bf16)qv[j];
    }
    __syncthreads();

    // --- encoder s ---------------------------------------------------------
    wave_gemm<0>(bufB, M, bufA, H, gs_w_in, gs_b_in, nullptr, nullptr, nullptr, 0, tid);
    __syncthreads();
#pragma unroll
    for (int i = 0; i < R; ++i) {
        wave_gemm<0>(bufA, H, bufB, H, gs_res_w + (size_t)i * H * H, gs_res_b + i * H,
                     nullptr, nullptr, nullptr, 0, tid);
        __syncthreads();
        ln_relu_add(bufA, bufB, gs_ln_g + i * H, gs_ln_b + i * H, tid);
        __syncthreads();
    }
    // final: x_hat = scale * (h @ W_out^T + b) + shift -> global f32
    wave_gemm<2>(bufA, H, bufB, D, gs_w_out, gs_b_out, scale, shift, out, growBase, tid);
}

// ---------------------------------------------------------------------------
// Host launcher. Inputs (setup_inputs order):
//  0 x, 1 scale, 2 shift,
//  3 ga_w_in, 4 ga_b_in, 5 ga_res_w, 6 ga_res_b, 7 ga_ln_g, 8 ga_ln_b,
//  9 ga_w_out, 10 ga_b_out,
// 11 gs_w_in, 12 gs_b_in, 13 gs_res_w, 14 gs_res_b, 15 gs_ln_g, 16 gs_ln_b,
// 17 gs_w_out, 18 gs_b_out
// ---------------------------------------------------------------------------
extern "C" void kernel_launch(void* const* d_in, const int* in_sizes, int n_in,
                              void* d_out, int out_size, void* d_ws, size_t ws_size,
                              hipStream_t stream)
{
    (void)in_sizes; (void)n_in; (void)out_size; (void)ws_size;

    const float* x     = (const float*)d_in[0];
    const float* scale = (const float*)d_in[1];
    const float* shift = (const float*)d_in[2];

    // bf16 weight workspace layout (elements)
    bf16* ws       = (bf16*)d_ws;
    bf16* w_ga_in  = ws;                                  // H*D   = 65536
    bf16* w_ga_res = w_ga_in  + (size_t)H * D;            // R*H*H = 524288
    bf16* w_ga_out = w_ga_res + (size_t)R * H * H;        // M*H   = 131072
    bf16* w_gs_in  = w_ga_out + (size_t)M * H;            // H*M   = 131072
    bf16* w_gs_res = w_gs_in  + (size_t)H * M;            // R*H*H = 524288
    bf16* w_gs_out = w_gs_res + (size_t)R * H * H;        // D*H   = 65536

    const struct { const float* src; bf16* dst; int n; } cv[6] = {
        { (const float*)d_in[3],  w_ga_in,  H * D },
        { (const float*)d_in[5],  w_ga_res, R * H * H },
        { (const float*)d_in[9],  w_ga_out, M * H },
        { (const float*)d_in[11], w_gs_in,  H * M },
        { (const float*)d_in[13], w_gs_res, R * H * H },
        { (const float*)d_in[17], w_gs_out, D * H },
    };
    for (int i = 0; i < 6; ++i) {
        cvt_f32_bf16<<<(cv[i].n + 255) / 256, 256, 0, stream>>>(cv[i].src, cv[i].dst, cv[i].n);
    }

    (void)hipFuncSetAttribute((const void*)fused_nwc_kernel,
                              hipFuncAttributeMaxDynamicSharedMemorySize, SMEM_BYTES);

    fused_nwc_kernel<<<N_ROWS / TILE, BLOCK, SMEM_BYTES, stream>>>(
        x, scale, shift,
        w_ga_in,  (const float*)d_in[4],
        w_ga_res, (const float*)d_in[6],  (const float*)d_in[7],  (const float*)d_in[8],
        w_ga_out, (const float*)d_in[10],
        w_gs_in,  (const float*)d_in[12],
        w_gs_res, (const float*)d_in[14], (const float*)d_in[15], (const float*)d_in[16],
        w_gs_out, (const float*)d_in[18],
        (float*)d_out);
}
